// HedgeDogConvAttention_27023934226683
// MI455X (gfx1250) — matrix operand order repaired and checked by reading, not compiled, not run on hardware
//
#include <hip/hip_runtime.h>
#include <hip/hip_bf16.h>

// HedgeHog conv linear-attention for MI455X (gfx1250, wave32, WMMA + TDM).
// out = Qf (Kf^T Vf) / (Qf . sum_n kf + eps)  -- exact refactoring of the
// reference quadratic form (associativity only): ~2.3 GFLOP, ~25 MB traffic,
// memory-bound at 23.3 TB/s. All GEMM stages on v_wmma_f32_16x16x32_f16.

typedef __attribute__((ext_vector_type(16))) _Float16 v16h;
typedef __attribute__((ext_vector_type(8)))  _Float16 v8h;
typedef __attribute__((ext_vector_type(8)))  float    v8f;
typedef __attribute__((ext_vector_type(4)))  unsigned u32x4;
typedef __attribute__((ext_vector_type(8)))  int      i32x8;
typedef __attribute__((ext_vector_type(4)))  int      i32x4;

#define NTOK 4096
#define LDS_PAD 40   // 80B pitch: 64B tile row + 16B pad; rows stay 16B-aligned
#define HOFF ((size_t)NTOK * 64)

// ---- WMMA fragment loads (CDNA5 ISA 7.12.2) as wide LDS reads ----
__device__ __forceinline__ v16h frag_a(const _Float16 (*As)[LDS_PAD], int lane, int wm) {
  const int m = wm + (lane & 15);
  const int o = (lane & 16) ? 8 : 0;
  v8h lo = *(const v8h*)(&As[m][o]);
  v8h hi = *(const v8h*)(&As[m][16 + o]);
  return __builtin_shufflevector(lo, hi, 0, 1, 2, 3, 4, 5, 6, 7,
                                 8, 9, 10, 11, 12, 13, 14, 15);
}
__device__ __forceinline__ v16h frag_b(const _Float16 (*Bt)[LDS_PAD], int lane, int wn) {
  const int n = wn + (lane & 15);
  const int k0 = (lane & 16) ? 8 : 0;
  v8h lo = *(const v8h*)(&Bt[n][k0]);
  v8h hi = *(const v8h*)(&Bt[n][k0 + 16]);
  return __builtin_shufflevector(lo, hi, 0, 1, 2, 3, 4, 5, 6, 7,
                                 8, 9, 10, 11, 12, 13, 14, 15);
}

// 8 f32 -> 8 f16 via two global_load_b128
__device__ __forceinline__ v8h cvt8(const float* __restrict__ p) {
  const float4 f0 = *(const float4*)p;
  const float4 f1 = *(const float4*)(p + 4);
  v8h h;
  h[0] = (_Float16)f0.x; h[1] = (_Float16)f0.y;
  h[2] = (_Float16)f0.z; h[3] = (_Float16)f0.w;
  h[4] = (_Float16)f1.x; h[5] = (_Float16)f1.y;
  h[6] = (_Float16)f1.z; h[7] = (_Float16)f1.w;
  return h;
}

// ---- Tensor Data Mover: 2D f16 tile -> pitched LDS tile (ISA §8.3/8.4) ----
__device__ __forceinline__ unsigned lds_addr_of(const void* p) {
  return (unsigned)(uintptr_t)(__attribute__((address_space(3))) const void*)p;
}
__device__ __forceinline__ void tdm_load_2d_f16(unsigned lds_addr, const void* gaddr,
                                                unsigned tensor_d0, unsigned tensor_d1,
                                                unsigned tile_d0, unsigned tile_d1,
                                                unsigned stride0_elems) {
  const unsigned long long ga = (unsigned long long)(uintptr_t)gaddr;
  u32x4 g0;
  g0[0] = 1u;                                   // count=1 (valid user descriptor)
  g0[1] = lds_addr;                             // LDS byte address
  g0[2] = (unsigned)ga;                         // global_addr[31:0]
  g0[3] = (unsigned)((ga >> 32) & 0x01FFFFFFu) | 0x80000000u; // addr[56:32], type=2
  i32x8 g1;
  g1[0] = (1 << 16)        // data_size = 2 bytes
        | (1 << 20)        // pad_enable
        | (3 << 22)        // pad_interval: every 16 DWORDs (64B row)
        | (3 << 25);       // pad_amount: 4 DWORDs (16B) -> 80B pitch
  g1[1] = (int)((tensor_d0 & 0xFFFFu) << 16);
  g1[2] = (int)(((tensor_d0 >> 16) & 0xFFFFu) | ((tensor_d1 & 0xFFFFu) << 16));
  g1[3] = (int)(((tensor_d1 >> 16) & 0xFFFFu) | ((tile_d0 & 0xFFFFu) << 16));
  g1[4] = (int)(tile_d1 & 0xFFFFu);
  g1[5] = (int)stride0_elems;
  g1[6] = 0; g1[7] = 0;
  i32x4 z4 = {0, 0, 0, 0};
#if __has_include(<hip/amd_detail/amd_gfx1250_TDM.h>)
  i32x8 z8 = {0, 0, 0, 0, 0, 0, 0, 0};
  __builtin_amdgcn_tensor_load_to_lds(g0, g1, z4, z4, z8, 0);
#else
  __builtin_amdgcn_tensor_load_to_lds(g0, g1, z4, z4, 0);
#endif
}

// ============ 1) q/k 1x1x1 projections: [256,256] x [256,4096] ============
__global__ __launch_bounds__(128) void proj_qk_kernel(
    const float* __restrict__ Wq, const float* __restrict__ Wk,
    const float* __restrict__ x, float* __restrict__ q, float* __restrict__ k) {
  const float* __restrict__ W  = blockIdx.z ? Wk : Wq;
  float* __restrict__ out      = blockIdx.z ? k  : q;
  const int nBase = blockIdx.x * 32, mBase = blockIdx.y * 32;
  __shared__ __align__(16) _Float16 As[32][LDS_PAD];  // [m][k]
  __shared__ __align__(16) _Float16 Bt[32][LDS_PAD];  // [n][k]
  const int t = threadIdx.x, lane = t & 31, wave = t >> 5;
  const int wm = (wave >> 1) * 16, wn = (wave & 1) * 16;
  const int r = t >> 2, seg = (t & 3) * 8;     // 8 contiguous elems per thread
  const float* pW = &W[(mBase + r) * 256 + seg];
  const float* pX = &x[(size_t)r * NTOK + nBase + seg];
  v8f acc = {};
  for (int kc = 0; kc < 256; kc += 32) {
    *(v8h*)&As[r][seg] = cvt8(pW);             // one ds_store_b128
    v8h bh = cvt8(pX);
#pragma unroll
    for (int j = 0; j < 8; ++j) Bt[seg + j][r] = bh[j];  // transposed scatter
    pW += 32;
    pX += (size_t)32 * NTOK;
    __syncthreads();
    if (kc + 32 < 256) __builtin_prefetch(pX, 0, 1);
    v16h a = frag_a(As, lane, wm), b = frag_b(Bt, lane, wn);
    acc = __builtin_amdgcn_wmma_f32_16x16x32_f16(false, a, false, b,
                                                 (short)0, acc, false, false);
    __syncthreads();
  }
  const int col  = nBase + wn + (lane & 15);
  const int row0 = mBase + wm + ((lane & 16) ? 8 : 0);
#pragma unroll
  for (int rr = 0; rr < 8; ++rr) out[(size_t)(row0 + rr) * NTOK + col] = acc[rr];
}

// ============ 2) depthwise 3x3x3 conv + bias -> v [256,4096] ============
__global__ __launch_bounds__(256) void convv_kernel(
    const float* __restrict__ x, const float* __restrict__ Wv,
    const float* __restrict__ bv, float* __restrict__ v) {
  const int idx = blockIdx.x * 256 + threadIdx.x;
  const int c = idx >> 12, n = idx & 4095;
  const int a0 = n >> 8, a1 = (n >> 4) & 15, a2 = n & 15;
  const float* __restrict__ xc = x + (size_t)c * 4096;
  const float* __restrict__ wc = Wv + c * 27;
  float s = bv[c];
#pragma unroll
  for (int i = -1; i <= 1; ++i) {
    const int p0 = a0 + i;
    if ((unsigned)p0 >= 16u) continue;
#pragma unroll
    for (int j = -1; j <= 1; ++j) {
      const int p1 = a1 + j;
      if ((unsigned)p1 >= 16u) continue;
#pragma unroll
      for (int kk = -1; kk <= 1; ++kk) {
        const int p2 = a2 + kk;
        if ((unsigned)p2 >= 16u) continue;
        s += xc[p0 * 256 + p1 * 16 + p2] * wc[(i + 1) * 9 + (j + 1) * 3 + (kk + 1)];
      }
    }
  }
  v[idx] = s;
}

// ============ 3) hedgehog: exp(+-(z Wm^T + b)) -> f16 [8][4096][64] ============
__global__ __launch_bounds__(128) void hedgehog_kernel(
    const float* __restrict__ src, const float* __restrict__ Wm,
    const float* __restrict__ bm, _Float16* __restrict__ dst) {
  const int nb = blockIdx.x * 32, eb = blockIdx.y * 32, h = blockIdx.z;
  __shared__ __align__(16) _Float16 As[32][LDS_PAD];  // [token][d]
  __shared__ __align__(16) _Float16 Bt[32][LDS_PAD];  // [e][d]
  const int t = threadIdx.x, lane = t & 31, wave = t >> 5;
  const int wm = (wave >> 1) * 16, wn = (wave & 1) * 16;
  const int r = t >> 2, seg = (t & 3) * 8;
  const float* pS = &src[(size_t)(h * 64 + r) * NTOK + nb + seg];  // r = d row
  const float* pB = &Wm[(eb + r) * 64 + seg];                      // r = e row
  v8f acc = {};
  for (int d0 = 0; d0 < 64; d0 += 32) {
    v8h ah = cvt8(pS);                         // 8 tokens of one d-row
#pragma unroll
    for (int j = 0; j < 8; ++j) As[seg + j][r] = ah[j];  // [token][d] scatter
    *(v8h*)&Bt[r][seg] = cvt8(pB);             // contiguous d: ds_store_b128
    pS += (size_t)32 * NTOK;
    pB += 32;
    __syncthreads();
    v16h a = frag_a(As, lane, wm), b = frag_b(Bt, lane, wn);
    acc = __builtin_amdgcn_wmma_f32_16x16x32_f16(false, a, false, b,
                                                 (short)0, acc, false, false);
    __syncthreads();
  }
  const int e    = eb + wn + (lane & 15);
  const int tok0 = nb + wm + ((lane & 16) ? 8 : 0);
  const float bias = bm[e];
#pragma unroll
  for (int rr = 0; rr < 8; ++rr) {
    const float hv = acc[rr] + bias;
    const int tok  = tok0 + rr;
    dst[(size_t)h * HOFF + (size_t)tok * 64 + e]       = (_Float16)__expf(hv);
    dst[(size_t)(h + 4) * HOFF + (size_t)tok * 64 + e] = (_Float16)__expf(-hv);
  }
}

// ============ 4) KV = Kf^T Vf per head, 8-way split-K + f32 atomics ============
#define KVSPLIT 8
__global__ __launch_bounds__(128) void kv_kernel(
    const _Float16* __restrict__ kf, const _Float16* __restrict__ vf,
    float* __restrict__ KV) {
  const int db = (blockIdx.x >> 3) * 32;
  const int sp = blockIdx.x & (KVSPLIT - 1);
  const int eb = blockIdx.y * 32;
  const int h  = blockIdx.z;
  __shared__ __align__(16) _Float16 As[32][LDS_PAD];  // [d][n]
  __shared__ __align__(16) _Float16 Bt[32][LDS_PAD];  // [e][n]
  const int t = threadIdx.x, lane = t & 31, wave = t >> 5;
  const int wm = (wave >> 1) * 16, wn = (wave & 1) * 16;
  const int r = t >> 2, seg = (t & 3) * 8;    // r = token row within tile
  const int nBeg = sp * (NTOK / KVSPLIT);
  const _Float16* pK = kf + (size_t)h * HOFF + (size_t)(nBeg + r) * 64 + db + seg;
  const _Float16* pV = vf + (size_t)h * HOFF + (size_t)(nBeg + r) * 64 + eb + seg;
  v8f acc = {};
  for (int it = 0; it < (NTOK / KVSPLIT) / 32; ++it) {
    v8h ah = *(const v8h*)pK;                  // one global_load_b128 (f16)
    v8h bh = *(const v8h*)pV;
#pragma unroll
    for (int j = 0; j < 8; ++j) {
      As[seg + j][r] = ah[j];                  // [d][n] transposed scatter
      Bt[seg + j][r] = bh[j];                  // [e][n] transposed scatter
    }
    pK += 32 * 64;
    pV += 32 * 64;
    __syncthreads();
    if (it + 1 < (NTOK / KVSPLIT) / 32) {
      __builtin_prefetch(pK, 0, 1);
      __builtin_prefetch(pV, 0, 1);
    }
    v16h a = frag_a(As, lane, wm), b = frag_b(Bt, lane, wn);
    acc = __builtin_amdgcn_wmma_f32_16x16x32_f16(false, a, false, b,
                                                 (short)0, acc, false, false);
    __syncthreads();
  }
  const int e  = eb + wn + (lane & 15);
  const int d0 = db + wm + ((lane & 16) ? 8 : 0);
#pragma unroll
  for (int rr = 0; rr < 8; ++rr)
    atomicAdd(&KV[(size_t)h * 4096 + (size_t)(d0 + rr) * 64 + e], acc[rr]);
}

__global__ __launch_bounds__(256) void zero_kernel(float* __restrict__ p, int n) {
  const int i = blockIdx.x * 256 + threadIdx.x;
  if (i < n) p[i] = 0.0f;
}

// ============ 5) ksum[h][d] = sum_n kf[h][n][d] ============
__global__ __launch_bounds__(256) void ksum_kernel(const _Float16* __restrict__ kf,
                                                   float* __restrict__ ksum) {
  const int h = blockIdx.x;
  const _Float16* __restrict__ kh = kf + (size_t)h * HOFF;
  __shared__ float red[256];
  const int t = threadIdx.x;
  const int d = t & 63, part = t >> 6;
  float s = 0.0f;
  const int nb = part * (NTOK / 4);
  for (int n = nb; n < nb + NTOK / 4; ++n) s += (float)kh[(size_t)n * 64 + d];
  red[t] = s;
  __syncthreads();
  if (t < 64) ksum[h * 64 + t] = red[t] + red[t + 64] + red[t + 128] + red[t + 192];
}

// ============ 6) out = (Qf KV) / (Qf ksum + eps); A tile fetched by TDM ============
__global__ __launch_bounds__(128) void out_kernel(
    const _Float16* __restrict__ qf, const float* __restrict__ KV,
    const float* __restrict__ ksum, float* __restrict__ out) {
  const int nb = blockIdx.x * 32, eb = blockIdx.y * 32, h = blockIdx.z;
  const _Float16* __restrict__ qh = qf + (size_t)h * HOFF;
  const float* __restrict__ KVh = KV + (size_t)h * 4096;
  const float* __restrict__ ksh = ksum + h * 64;
  __shared__ __align__(16) _Float16 As[32][LDS_PAD];  // [token][d], TDM-filled
  __shared__ __align__(16) _Float16 Bt[32][LDS_PAD];  // [e][d]
  __shared__ float den[32];
  const int t = threadIdx.x, lane = t & 31, wave = t >> 5;
  const int wm = (wave >> 1) * 16, wn = (wave & 1) * 16;
  const int r = t >> 2, seg = (t & 3) * 8;
  if (t < 32) {  // denominator: qf[n] . ksum + eps (wide f16 loads)
    float s = 1e-6f;
    const _Float16* pq = &qh[(size_t)(nb + t) * 64];
#pragma unroll
    for (int d = 0; d < 64; d += 8) {
      v8h qv = *(const v8h*)&pq[d];
#pragma unroll
      for (int j = 0; j < 8; ++j) s += (float)qv[j] * ksh[d + j];
    }
    den[t] = s;
  }
  v8f acc = {};
  const unsigned asLds = lds_addr_of(&As[0][0]);
  const float* pKV = &KVh[(size_t)r * 64 + eb + seg];   // r = d row
  for (int d0 = 0; d0 < 64; d0 += 32) {
    if (wave == 0) {  // TDM: 32x32 f16 tile, row stride 64 elems, 80B LDS pitch
      tdm_load_2d_f16(asLds, &qh[(size_t)nb * 64 + d0],
                      /*tensor_d0=*/64, /*tensor_d1=*/NTOK,
                      /*tile_d0=*/32, /*tile_d1=*/32, /*stride0=*/64);
      __builtin_amdgcn_s_wait_tensorcnt((short)0);
    }
    v8h bh = cvt8(pKV);
#pragma unroll
    for (int j = 0; j < 8; ++j) Bt[seg + j][r] = bh[j];  // [e][d] scatter
    pKV += 32 * 64;
    __syncthreads();
    v16h a = frag_a(As, lane, wm), b = frag_b(Bt, lane, wn);
    acc = __builtin_amdgcn_wmma_f32_16x16x32_f16(false, a, false, b,
                                                 (short)0, acc, false, false);
    __syncthreads();
  }
  const int e     = eb + wn + (lane & 15);
  const int tokl0 = wm + ((lane & 16) ? 8 : 0);
#pragma unroll
  for (int rr = 0; rr < 8; ++rr) {
    const int tokl = tokl0 + rr;
    out[(size_t)(h * 64 + e) * NTOK + nb + tokl] = acc[rr] / den[tokl];
  }
}

extern "C" void kernel_launch(void* const* d_in, const int* in_sizes, int n_in,
                              void* d_out, int out_size, void* d_ws, size_t ws_size,
                              hipStream_t stream) {
  (void)in_sizes; (void)n_in; (void)out_size; (void)ws_size;
  const float* x   = (const float*)d_in[0];
  const float* Wq  = (const float*)d_in[1];
  const float* Wk  = (const float*)d_in[2];
  const float* Wv  = (const float*)d_in[3];
  const float* bv  = (const float*)d_in[4];
  const float* Wmq = (const float*)d_in[5];
  const float* bmq = (const float*)d_in[6];
  const float* Wmk = (const float*)d_in[7];
  const float* bmk = (const float*)d_in[8];
  const float* Wmv = (const float*)d_in[9];
  const float* bmv = (const float*)d_in[10];
  float* out = (float*)d_out;

  float* q = (float*)d_ws;
  float* k = q + (size_t)256 * NTOK;
  float* v = k + (size_t)256 * NTOK;
  _Float16* qf = (_Float16*)(v + (size_t)256 * NTOK);
  _Float16* kf = qf + 8 * HOFF;
  _Float16* vf = kf + 8 * HOFF;
  float* KV   = (float*)(vf + 8 * HOFF);
  float* ksum = KV + 8 * 64 * 64;

  proj_qk_kernel<<<dim3(NTOK / 32, 8, 2), 128, 0, stream>>>(Wq, Wk, x, q, k);
  convv_kernel<<<(256 * NTOK) / 256, 256, 0, stream>>>(x, Wv, bv, v);
  hedgehog_kernel<<<dim3(NTOK / 32, 2, 4), 128, 0, stream>>>(q, Wmq, bmq, qf);
  hedgehog_kernel<<<dim3(NTOK / 32, 2, 4), 128, 0, stream>>>(k, Wmk, bmk, kf);
  hedgehog_kernel<<<dim3(NTOK / 32, 2, 4), 128, 0, stream>>>(v, Wmv, bmv, vf);
  zero_kernel<<<(8 * 64 * 64 + 255) / 256, 256, 0, stream>>>(KV, 8 * 64 * 64);
  kv_kernel<<<dim3(2 * KVSPLIT, 2, 8), 128, 0, stream>>>(kf, vf, KV);
  ksum_kernel<<<8, 256, 0, stream>>>(kf, ksum);
  out_kernel<<<dim3(NTOK / 32, 2, 8), 128, 0, stream>>>(qf, KV, ksum, out);
}
